// EntropyRegularizedVQ_23536420782642
// MI455X (gfx1250) — matrix operand order repaired and checked by reading, not compiled, not run on hardware
//
#include <hip/hip_runtime.h>
#include <hip/hip_bf16.h>
#include <math.h>

// ---------------------------------------------------------------------------
// EntropyRegularizedVQ for MI455X (gfx1250, wave32, WMMA)
// d[n,k] = ||z_n||^2 + ||w_k||^2 - 2 z_n.w_k ; argmin_k fused into GEMM.
// bf16 hi/lo split (3 WMMAs, 3 independent accumulator chains) for
// ~fp32-accurate dot products on the matrix cores.
// B tiles double-buffered in LDS via GLOBAL_LOAD_ASYNC_TO_LDS_B128 (ASYNCcnt).
// ---------------------------------------------------------------------------

typedef __bf16 v16bf __attribute__((ext_vector_type(16)));
typedef float  v8f   __attribute__((ext_vector_type(8)));

#define N_TOK 32768
#define K_CB  8192
#define DFEAT 128      // 2*DIM
#define DIMC  64

// workspace byte offsets
#define WS_A    0u          // 32768*128*2 bf16 (hi+lo, WMMA A-fragment layout) = 16 MB
#define WS_B    16777216u   //  8192*128*2 bf16 (hi+lo, WMMA B-fragment layout) =  4 MB
#define WS_YSQ  20971520u   //  8192 f32
#define WS_IDX  21004288u   // 32768 i32
#define WS_CNT  21135360u   //  8192 i32

// --- CDNA5 async global->LDS copy (falls back to load+ds_store) ------------
#if defined(__has_builtin)
#  if __has_builtin(__builtin_amdgcn_global_load_async_to_lds_b128)
#    define HAVE_ASYNC_LDS 1
#  endif
#endif
#ifndef HAVE_ASYNC_LDS
#  define HAVE_ASYNC_LDS 0
#endif

#if HAVE_ASYNC_LDS
// Builtin signature (from hipcc diagnostic): param is pointer to
// __attribute__((vector_size(16))) int, AS1 (printed '__device__') for the
// global source; AS3 for the LDS destination.
typedef int av4i __attribute__((vector_size(16)));
typedef __attribute__((address_space(1))) av4i g_av4i;
typedef __attribute__((address_space(3))) av4i l_av4i;
#  define ASYNC_COPY_B128(gsrc, ldst) \
     __builtin_amdgcn_global_load_async_to_lds_b128((g_av4i*)(gsrc), (l_av4i*)(ldst), 0, 0)
#  if __has_builtin(__builtin_amdgcn_s_wait_asynccnt)
#    define WAIT_ASYNC() __builtin_amdgcn_s_wait_asynccnt(0)
#  else
#    define WAIT_ASYNC() asm volatile("s_wait_asynccnt 0x0" ::: "memory")
#  endif
#endif

// ---- Stage z (tokens) into A-fragment layout, bf16 hi/lo -------------------
// A 16x32 bf16 fragment (ISA 7.12.2): lane L holds row M=L&15;
// VGPR v holds K = 2v + (v>=4 ? 8 : 0) + 8*(L>=16) + {0,1}.
__global__ void vq_prep_z(const float* __restrict__ zr, const float* __restrict__ zi,
                          __bf16* __restrict__ wsA) {
  unsigned gid = blockIdx.x * 256u + threadIdx.x;      // < 8,388,608
  unsigned j    = gid & 15u;
  unsigned lane = (gid >> 4) & 31u;
  unsigned h    = (gid >> 9) & 1u;                     // 0=hi, 1=lo
  unsigned ks   = (gid >> 10) & 3u;                    // k-step (32 feats each)
  unsigned mb   = gid >> 12;                           // 16-row block
  unsigned v  = j >> 1;
  unsigned kl = 2u * v + (v >= 4u ? 8u : 0u) + ((lane >= 16u) ? 8u : 0u) + (j & 1u);
  unsigned f  = ks * 32u + kl;
  unsigned row = mb * 16u + (lane & 15u);
  float x = (f < 64u) ? zr[row * 64u + f] : zi[row * 64u + (f - 64u)];
  __bf16 hi = (__bf16)x;
  wsA[gid] = (h == 0u) ? hi : (__bf16)(x - (float)hi);
}

// ---- Stage weights into B-fragment layout, bf16 hi/lo ----------------------
// B 32x16 bf16 fragment: lane L holds column N=L&15; element j -> K = j + 8*(j>=8) + 8*(L>=16).
__global__ void vq_prep_w(const float* __restrict__ w, __bf16* __restrict__ wsB) {
  unsigned gid = blockIdx.x * 256u + threadIdx.x;      // < 2,097,152
  unsigned j    = gid & 15u;
  unsigned lane = (gid >> 4) & 31u;
  unsigned h    = (gid >> 9) & 1u;
  unsigned ks   = (gid >> 10) & 3u;
  unsigned cb   = gid >> 12;                           // 16-concept block
  unsigned f   = ks * 32u + j + ((j >= 8u) ? 8u : 0u) + ((lane >= 16u) ? 8u : 0u);
  unsigned col = cb * 16u + (lane & 15u);
  float x = w[col * 128u + f];
  __bf16 hi = (__bf16)x;
  wsB[gid] = (h == 0u) ? hi : (__bf16)(x - (float)hi);
}

__global__ void vq_prep_ysq(const float* __restrict__ w, float* __restrict__ ysq) {
  int k = blockIdx.x * 256 + threadIdx.x;              // < 8192
  const float* row = w + (size_t)k * 128;
  float s = 0.f;
#pragma unroll 8
  for (int f = 0; f < 128; ++f) s += row[f] * row[f];
  ysq[k] = s;
}

__global__ void vq_zero_counts(int* __restrict__ cnt) {
  int k = blockIdx.x * 256 + threadIdx.x;
  if (k < K_CB) cnt[k] = 0;
}

// ---- Main fused GEMM + argmin ---------------------------------------------
// 256 WGs x 8 waves; wave handles rows [mb*16, mb*16+16), sweeps 512 col tiles.
// Double-buffered 8 KB B tiles in LDS; async copy of tile cb+1 overlapped with
// 12 WMMAs on tile cb; one barrier per tile.
__global__ __launch_bounds__(256, 1) void vq_argmin(const __bf16* __restrict__ wsA,
                                                    const __bf16* __restrict__ wsB,
                                                    const float* __restrict__ ysq,
                                                    int* __restrict__ outIdx) {
  __shared__ __align__(32) unsigned char lds[2][8192];
  const int tid  = threadIdx.x;
  const int wave = tid >> 5;
  const int lane = tid & 31;
  const int mb   = blockIdx.x * 8 + wave;              // 16-row block id

  // Persistent A fragments: 4 k-steps x {hi,lo} = 64 VGPRs
  const v16bf* Ap = (const v16bf*)wsA;
  v16bf a_hi[4], a_lo[4];
#pragma unroll
  for (int ks = 0; ks < 4; ++ks) {
    a_hi[ks] = Ap[(mb * 8 + ks * 2 + 0) * 32 + lane];
    a_lo[ks] = Ap[(mb * 8 + ks * 2 + 1) * 32 + lane];
  }

  float minv[8];
  int   mini[8];
#pragma unroll
  for (int v = 0; v < 8; ++v) { minv[v] = 3.4e38f; mini[v] = 0; }

  const uint4* Bsrc = (const uint4*)wsB;               // 16B granules; 512/tile

  // ---- prologue: stage tile 0 into buffer 0 ----
#if HAVE_ASYNC_LDS
  ASYNC_COPY_B128(Bsrc + tid,       lds[0] + tid * 16);
  ASYNC_COPY_B128(Bsrc + 256 + tid, lds[0] + (256 + tid) * 16);
  WAIT_ASYNC();
#else
  {
    uint4* Bl = (uint4*)lds[0];
    Bl[tid]       = Bsrc[tid];
    Bl[tid + 256] = Bsrc[256 + tid];
  }
#endif
  __syncthreads();

  float yv = ysq[lane & 15];                           // tile 0, pipelined

  for (int cb = 0; cb < 512; ++cb) {
    const int nb = (cb + 1) & 1;

    // ---- issue copy of next tile (overlaps with WMMAs below) ----
    if (cb + 1 < 512) {
      const int nbase = (cb + 1) * 512;
#if HAVE_ASYNC_LDS
      ASYNC_COPY_B128(Bsrc + nbase + tid,       lds[nb] + tid * 16);
      ASYNC_COPY_B128(Bsrc + nbase + 256 + tid, lds[nb] + (256 + tid) * 16);
#else
      uint4* Bl = (uint4*)lds[nb];
      Bl[tid]       = Bsrc[nbase + tid];
      Bl[tid + 256] = Bsrc[nbase + 256 + tid];
      if (cb + 2 < 512)                                // -> global_prefetch_b8
        __builtin_prefetch((const void*)(Bsrc + nbase + 512 + tid), 0, 1);
#endif
    }
    const float yv_next = (cb + 1 < 512) ? ysq[(cb + 1) * 16 + (lane & 15)] : 0.f;

    // ---- 12 WMMAs on current tile, 3 independent accumulator chains ----
    const unsigned char* cw = lds[cb & 1];
    v8f acc_hh = {}, acc_hl = {}, acc_lh = {};
#pragma unroll
    for (int ks = 0; ks < 4; ++ks) {
      v16bf bhi = *(const v16bf*)(cw + ((ks * 2 + 0) * 32 + lane) * 32);
      v16bf blo = *(const v16bf*)(cw + ((ks * 2 + 1) * 32 + lane) * 32);
      acc_hh = __builtin_amdgcn_wmma_f32_16x16x32_bf16(false, a_hi[ks], false, bhi,
                                                       (short)0, acc_hh, false, false);
      acc_hl = __builtin_amdgcn_wmma_f32_16x16x32_bf16(false, a_hi[ks], false, blo,
                                                       (short)0, acc_hl, false, false);
      acc_lh = __builtin_amdgcn_wmma_f32_16x16x32_bf16(false, a_lo[ks], false, bhi,
                                                       (short)0, acc_lh, false, false);
    }

    // C/D layout: VGPR v -> row v (lanes 0-15) or v+8 (lanes 16-31); col = lane&15
    const int col = cb * 16 + (lane & 15);
#pragma unroll
    for (int v = 0; v < 8; ++v) {
      float d = yv - 2.0f * (acc_hh[v] + (acc_hl[v] + acc_lh[v]));
      if (d < minv[v]) { minv[v] = d; mini[v] = col; }
    }
    yv = yv_next;

#if HAVE_ASYNC_LDS
    WAIT_ASYNC();                                      // next tile landed in LDS
#endif
    __syncthreads();
  }

  // Butterfly argmin across the 16 lanes of each half (xor masks stay in-half)
#pragma unroll
  for (int m = 1; m <= 8; m <<= 1) {
#pragma unroll
    for (int v = 0; v < 8; ++v) {
      float ov = __shfl_xor(minv[v], m, 32);
      int   oi = __shfl_xor(mini[v], m, 32);
      if (ov < minv[v] || (ov == minv[v] && oi < mini[v])) { minv[v] = ov; mini[v] = oi; }
    }
  }
  if ((lane & 15) == 0) {
    const int rb = mb * 16 + ((lane >> 4) ? 8 : 0);
#pragma unroll
    for (int v = 0; v < 8; ++v) outIdx[rb + v] = mini[v];
  }
}

// ---- Gather + loss + complex output + histogram ---------------------------
__global__ void vq_finalize(const float* __restrict__ w, const float* __restrict__ zr,
                            const float* __restrict__ zi, const int* __restrict__ idx,
                            float* __restrict__ out, int* __restrict__ cnt) {
  const int n = blockIdx.x * 256 + threadIdx.x;        // < 32768
  const int k = idx[n];
  const float* wr = w + (size_t)k * 128;
  float* o = out + (size_t)n * 128;
  float acc = 0.f;
#pragma unroll 8
  for (int d = 0; d < 64; ++d) {
    float a = wr[d], b = wr[64 + d];
    o[2 * d]     = a;                                  // real
    o[2 * d + 1] = b;                                  // imag (complex64 interleave)
    float dr = a - zr[n * 64 + d];
    float di = b - zi[n * 64 + d];
    acc += dr * dr + di * di;
  }
  // mean((zq-z)^2) + 0.25*mean((zq-z)^2): stop_gradient is identity in forward
  out[(size_t)N_TOK * 128 + n]         = 1.25f * acc * (1.0f / 128.0f);
  out[(size_t)N_TOK * 128 + N_TOK + n] = (float)k;
  atomicAdd(&cnt[k], 1);
}

__global__ void vq_entropy(const int* __restrict__ cnt, float* __restrict__ out) {
  __shared__ float red[256];
  const int t = threadIdx.x;
  float s = 0.f;
  for (int k = t; k < K_CB; k += 256) {
    float p = (float)cnt[k] * (1.0f / (float)N_TOK);
    s += p * logf(p + 1e-10f);                         // p==0 contributes exactly 0
  }
  red[t] = s;
  __syncthreads();
  for (int o = 128; o > 0; o >>= 1) {
    if (t < o) red[t] += red[t + o];
    __syncthreads();
  }
  if (t == 0) out[(size_t)N_TOK * 128 + 2 * N_TOK] = -red[0];
}

// ---------------------------------------------------------------------------
extern "C" void kernel_launch(void* const* d_in, const int* in_sizes, int n_in,
                              void* d_out, int out_size, void* d_ws, size_t ws_size,
                              hipStream_t stream) {
  const float* z_real = (const float*)d_in[0];
  const float* z_imag = (const float*)d_in[1];
  const float* weight = (const float*)d_in[2];
  float* out = (float*)d_out;
  char* ws = (char*)d_ws;

  __bf16* wsA  = (__bf16*)(ws + WS_A);
  __bf16* wsB  = (__bf16*)(ws + WS_B);
  float*  ysq  = (float*)(ws + WS_YSQ);
  int*    idx  = (int*)(ws + WS_IDX);
  int*    cnt  = (int*)(ws + WS_CNT);

  vq_prep_z<<<32768, 256, 0, stream>>>(z_real, z_imag, wsA);   // 8.39M elems
  vq_prep_w<<<8192, 256, 0, stream>>>(weight, wsB);            // 2.10M elems
  vq_prep_ysq<<<32, 256, 0, stream>>>(weight, ysq);
  vq_zero_counts<<<32, 256, 0, stream>>>(cnt);
  vq_argmin<<<256, 256, 0, stream>>>(wsA, wsB, ysq, idx);      // the GEMM+argmin
  vq_finalize<<<128, 256, 0, stream>>>(weight, z_real, z_imag, idx, out, cnt);
  vq_entropy<<<1, 256, 0, stream>>>(cnt, out);
  (void)in_sizes; (void)n_in; (void)out_size; (void)ws_size;
}
